// TinyCKModel_5025111736662
// MI455X (gfx1250) — compile-verified, matrix operand order checked by reading, not run on hardware
//
#include <hip/hip_runtime.h>

#define VOCAB  32000
#define DMODEL 1024
#define HIDDEN 4096
#define NTOK   4096      // B*S = 2*2048
#define EPSF   1e-6f
#define BK     32
#define LDK    40        // padded LDS row (bf16 elems) -> 80B, keeps 16B align, avoids bank conflicts

typedef __attribute__((ext_vector_type(16))) __bf16    v16bf;
typedef __attribute__((ext_vector_type(8)))  float     v8f;
typedef __attribute__((ext_vector_type(4)))  unsigned  uv4;   // native vector: OK for asm 'v' constraint

union Frag16 { uint4 u[2]; v16bf v; };

// f32 -> bf16, full round-to-nearest-even (epilogue / scattered stores only)
__device__ __forceinline__ unsigned short f2bf(float x) {
  unsigned u = __float_as_uint(x);
  u += 0x7FFFu + ((u >> 16) & 1u);
  return (unsigned short)(u >> 16);
}
// Pack two f32 -> two bf16 in one dword: round-half-up (+0x8000) then a single
// v_perm_b32 grabs the high halves.
__device__ __forceinline__ unsigned pack2(float a, float b) {
  unsigned ua = __float_as_uint(a) + 0x8000u;
  unsigned ub = __float_as_uint(b) + 0x8000u;
  // dst bytes: [0]=ua.b2 [1]=ua.b3 [2]=ub.b2 [3]=ub.b3  (sel 0-3 -> s1, 4-7 -> s0)
  return __builtin_amdgcn_perm(ub, ua, 0x07060302u);
}

// Async copy global -> LDS (raw bf16), tracked on ASYNCcnt.
// INST_OFFSET applies to both the LDS and the global address (ISA 08 §4.4),
// so one (lds,global) address pair covers all chunks of a contiguous row.
__device__ __forceinline__ void async_copy_32B(unsigned lds_off, unsigned long long gaddr) {
  asm volatile(
      "global_load_async_to_lds_b128 %0, %1, off\n\t"
      "global_load_async_to_lds_b128 %0, %1, off offset:16"
      :: "v"(lds_off), "v"(gaddr) : "memory");
}
__device__ __forceinline__ void async_copy_64B(unsigned lds_off, unsigned long long gaddr) {
  asm volatile(
      "global_load_async_to_lds_b128 %0, %1, off\n\t"
      "global_load_async_to_lds_b128 %0, %1, off offset:16\n\t"
      "global_load_async_to_lds_b128 %0, %1, off offset:32\n\t"
      "global_load_async_to_lds_b128 %0, %1, off offset:48"
      :: "v"(lds_off), "v"(gaddr) : "memory");
}
// Guaranteed-vector LDS store of 32B (2 x ds_store_b128); address is an LDS byte offset.
__device__ __forceinline__ void lds_store_32B(unsigned lds_off, uv4 a, uv4 b) {
  asm volatile(
      "ds_store_b128 %0, %1\n\t"
      "ds_store_b128 %0, %2 offset:16"
      :: "v"(lds_off), "v"(a), "v"(b) : "memory");
}
// Pre-barrier wait: asm DS stores + async loads must be complete/visible.
__device__ __forceinline__ void tile_waits() {
  asm volatile("s_wait_dscnt 0x0\n\ts_wait_asynccnt 0x0" ::: "memory");
}

// Load a 16x32 bf16 operand fragment (A row-major [m][k] or B stored [n][k])
// from LDS per the CDNA5 16-bit A-matrix lane layout:
//   lanes 0-15 : row = lane,    K chunks {0..7, 16..23}
//   lanes 16-31: row = lane-16, K chunks {8..15, 24..31}
__device__ __forceinline__ v16bf load_frag(const unsigned short* lds, int rowBase) {
  int lane = threadIdx.x & 31;
  int row  = lane & 15;
  int k0   = (lane >> 4) << 3;      // 0 or 8
  const unsigned short* p = lds + (rowBase + row) * LDK + k0;
  Frag16 f;
  f.u[0] = *(const uint4*)(p);      // K k0..k0+7
  f.u[1] = *(const uint4*)(p + 16); // K k0+16..k0+23
  return f.v;
}

__device__ __forceinline__ v8f wmma_bf16(v16bf a, v16bf b, v8f c) {
  return __builtin_amdgcn_wmma_f32_16x16x32_bf16(false, a, false, b, (short)0, c, false, false);
}

// Convert 16 f32 (4 float4) -> two packed native vectors of bf16
__device__ __forceinline__ void cvt16(float4 f0, float4 f1, float4 f2, float4 f3,
                                      uv4& p0, uv4& p1) {
  p0.x = pack2(f0.x, f0.y); p0.y = pack2(f0.z, f0.w);
  p0.z = pack2(f1.x, f1.y); p0.w = pack2(f1.z, f1.w);
  p1.x = pack2(f2.x, f2.y); p1.y = pack2(f2.z, f2.w);
  p1.z = pack2(f3.x, f3.y); p1.w = pack2(f3.z, f3.w);
}

// ---------------- Kernel 1: embedding gather + RMSNorm -> bf16 x ----------------
__global__ void __launch_bounds__(256)
embed_rmsnorm_kernel(const int* __restrict__ ids, const float* __restrict__ emb,
                     const float* __restrict__ gamma, unsigned short* __restrict__ xb) {
  __shared__ float red[256];
  int tok = blockIdx.x;
  int t   = threadIdx.x;
  int row = ids[tok];
  if (row < 0 || row >= VOCAB) row = 0;
  const float4 v = ((const float4*)(emb + (long)row * DMODEL))[t]; // 256*4 = 1024
  red[t] = v.x*v.x + v.y*v.y + v.z*v.z + v.w*v.w;
  __syncthreads();
  for (int s = 128; s > 0; s >>= 1) {
    if (t < s) red[t] += red[t + s];
    __syncthreads();
  }
  float rstd = rsqrtf(red[0] * (1.0f / DMODEL) + EPSF);
  const float4 g = ((const float4*)gamma)[t];
  uint2 o;
  o.x = pack2(v.x * rstd * g.x, v.y * rstd * g.y);
  o.y = pack2(v.z * rstd * g.z, v.w * rstd * g.w);
  *(uint2*)(xb + (long)tok * DMODEL + t * 4) = o;
}

// ---------------- Kernel 2: fc1 (bf16 WMMA) fused with SwiGLU -> bf16 y ----------------
// WG tile: 128 tokens x 64 hidden cols, computing both 'a' (w1 rows n) and
// 'g' (w1 rows HIDDEN+n) so silu(a)*g pairs live in matching accumulator lanes.
// Double-buffered LDS, async global->LDS for the bf16 A tiles.
#define G1_BM 128
#define G1_BN 64
__global__ void __launch_bounds__(256)
gemm1_swiglu_kernel(const unsigned short* __restrict__ xb, const float* __restrict__ w1,
                    const float* __restrict__ b1, unsigned short* __restrict__ yb) {
  __shared__ __align__(16) unsigned short As[2][G1_BM * LDK];
  __shared__ __align__(16) unsigned short Bs[2][G1_BM * LDK];  // rows 0-63: 'a', 64-127: 'g'
  int t    = threadIdx.x;
  int wave = t >> 5;
  int wm   = wave >> 1;                 // 0..3 -> 32-row slab
  int wn   = wave & 1;                  // 0..1 -> 32-col slab
  int mBase = blockIdx.x * G1_BM;
  int nBase = blockIdx.y * G1_BN;

  v8f accA[2][2] = {};
  v8f accG[2][2] = {};

  int lr = t >> 1;                      // 0..127: LDS tile row handled by this thread
  int lh = (t & 1) << 4;                // 0 or 16 (element offset within 32-wide K tile)
  const unsigned short* aRow = xb + (long)(mBase + lr) * DMODEL + lh;
  long w1r = (lr < 64) ? (long)(nBase + lr) : (long)(HIDDEN + nBase + (lr - 64));
  const float* bRow = w1 + w1r * DMODEL + lh;

  unsigned ldsA0 = (unsigned)(unsigned long long)&As[0][lr * LDK + lh];
  unsigned ldsA1 = (unsigned)(unsigned long long)&As[1][lr * LDK + lh];
  unsigned ldsB0 = (unsigned)(unsigned long long)&Bs[0][lr * LDK + lh];
  unsigned ldsB1 = (unsigned)(unsigned long long)&Bs[1][lr * LDK + lh];

  // ---- prologue: stage tile 0 into buffer 0 ----
  async_copy_32B(ldsA0, (unsigned long long)aRow);
  {
    float4 f0 = *(const float4*)(bRow);
    float4 f1 = *(const float4*)(bRow + 4);
    float4 f2 = *(const float4*)(bRow + 8);
    float4 f3 = *(const float4*)(bRow + 12);
    uv4 p0, p1;
    cvt16(f0, f1, f2, f3, p0, p1);
    lds_store_32B(ldsB0, p0, p1);
  }
  tile_waits();
  __syncthreads();

  const int STEPS = DMODEL / BK;        // 32
  for (int kk = 0; kk < STEPS; ++kk) {
    int cur = kk & 1;
    bool hasNext = (kk + 1) < STEPS;
    int knext = (kk + 1) * BK;
    float4 f0, f1, f2, f3;
    if (hasNext) {
      async_copy_32B(cur ? ldsA0 : ldsA1, (unsigned long long)(aRow + knext));
      f0 = *(const float4*)(bRow + knext);
      f1 = *(const float4*)(bRow + knext + 4);
      f2 = *(const float4*)(bRow + knext + 8);
      f3 = *(const float4*)(bRow + knext + 12);
      __builtin_prefetch(bRow + knext + BK, 0, 1);
    }

    // ---- compute tile kk from buffer cur (overlaps with loads above) ----
    const unsigned short* Ac = As[cur];
    const unsigned short* Bc = Bs[cur];
    v16bf fa[2], fba[2], fbg[2];
#pragma unroll
    for (int mt = 0; mt < 2; ++mt) fa[mt] = load_frag(Ac, wm * 32 + mt * 16);
#pragma unroll
    for (int nt = 0; nt < 2; ++nt) {
      fba[nt] = load_frag(Bc, wn * 32 + nt * 16);
      fbg[nt] = load_frag(Bc, 64 + wn * 32 + nt * 16);
    }
#pragma unroll
    for (int mt = 0; mt < 2; ++mt)
#pragma unroll
      for (int nt = 0; nt < 2; ++nt) {
        accA[mt][nt] = wmma_bf16(fa[mt], fba[nt], accA[mt][nt]);
        accG[mt][nt] = wmma_bf16(fa[mt], fbg[nt], accG[mt][nt]);
      }

    if (hasNext) {
      uv4 p0, p1;
      cvt16(f0, f1, f2, f3, p0, p1);
      lds_store_32B(cur ? ldsB0 : ldsB1, p0, p1);
    }
    tile_waits();
    __syncthreads();
  }

  // Epilogue: bias + SwiGLU entirely in registers, store bf16 y.
  int lane  = t & 31;
  int col   = lane & 15;
  int rbase = (lane >> 4) << 3;         // 0 or 8 per C-matrix layout
#pragma unroll
  for (int mt = 0; mt < 2; ++mt)
#pragma unroll
    for (int nt = 0; nt < 2; ++nt) {
      int n = nBase + wn * 32 + nt * 16 + col;
      float ba = b1[n];
      float bg = b1[HIDDEN + n];
#pragma unroll
      for (int v = 0; v < 8; ++v) {
        int m   = mBase + wm * 32 + mt * 16 + rbase + v;
        float a = accA[mt][nt][v] + ba;
        float g = accG[mt][nt][v] + bg;
        float y = (a / (1.0f + __expf(-a))) * g;  // silu(a) * g
        yb[(long)m * HIDDEN + n] = f2bf(y);
      }
    }
}

// ---------------- Kernel 3: fc2 (bf16 WMMA) -> f32 logits ----------------
// WG tile 256x128, 8 waves as 4(M)x2(N): wave tile 64x64 -> 16 WMMAs per
// 8 fragment loads per K-step. A tile (256x32 bf16) streamed via async
// global->LDS (zero VGPR staging); B tile (128x32 f32) converted on the fly.
#define G2_BM 256
#define G2_BN 128
__global__ void __launch_bounds__(256)
gemm2_kernel(const unsigned short* __restrict__ yb, const float* __restrict__ w2,
             const float* __restrict__ b2, float* __restrict__ out) {
  __shared__ __align__(16) unsigned short As[2][G2_BM * LDK];   // 2 x 20 KB
  __shared__ __align__(16) unsigned short Bs[2][G2_BN * LDK];   // 2 x 10 KB
  int t    = threadIdx.x;
  int wave = t >> 5;
  int wm   = wave >> 1;                 // 0..3 -> rows wm*64
  int wn   = wave & 1;                  // 0..1 -> cols wn*64
  int mBase = blockIdx.x * G2_BM;
  int nBase = blockIdx.y * G2_BN;

  v8f acc[4][4] = {};

  // A: one full 64B row per thread (async).  B: half row per thread (convert).
  const unsigned short* aRow = yb + (long)(mBase + t) * HIDDEN;
  int lr = t >> 1;
  int lh = (t & 1) << 4;
  const float* bRow = w2 + (long)(nBase + lr) * HIDDEN + lh;

  unsigned ldsA0 = (unsigned)(unsigned long long)&As[0][t * LDK];
  unsigned ldsA1 = (unsigned)(unsigned long long)&As[1][t * LDK];
  unsigned ldsB0 = (unsigned)(unsigned long long)&Bs[0][lr * LDK + lh];
  unsigned ldsB1 = (unsigned)(unsigned long long)&Bs[1][lr * LDK + lh];

  // ---- prologue: stage tile 0 ----
  async_copy_64B(ldsA0, (unsigned long long)aRow);
  {
    float4 f0 = *(const float4*)(bRow);
    float4 f1 = *(const float4*)(bRow + 4);
    float4 f2 = *(const float4*)(bRow + 8);
    float4 f3 = *(const float4*)(bRow + 12);
    uv4 p0, p1;
    cvt16(f0, f1, f2, f3, p0, p1);
    lds_store_32B(ldsB0, p0, p1);
  }
  tile_waits();
  __syncthreads();

  const int STEPS = HIDDEN / BK;        // 128
  for (int kk = 0; kk < STEPS; ++kk) {
    int cur = kk & 1;
    bool hasNext = (kk + 1) < STEPS;
    int knext = (kk + 1) * BK;
    float4 f0, f1, f2, f3;
    if (hasNext) {
      async_copy_64B(cur ? ldsA0 : ldsA1, (unsigned long long)(aRow + knext));
      f0 = *(const float4*)(bRow + knext);
      f1 = *(const float4*)(bRow + knext + 4);
      f2 = *(const float4*)(bRow + knext + 8);
      f3 = *(const float4*)(bRow + knext + 12);
      __builtin_prefetch(bRow + knext + BK, 0, 1);
    }

    const unsigned short* Ac = As[cur];
    const unsigned short* Bc = Bs[cur];
    v16bf fa[4];
#pragma unroll
    for (int mt = 0; mt < 4; ++mt) fa[mt] = load_frag(Ac, wm * 64 + mt * 16);
#pragma unroll
    for (int nt = 0; nt < 4; ++nt) {
      v16bf fb = load_frag(Bc, wn * 64 + nt * 16);
#pragma unroll
      for (int mt = 0; mt < 4; ++mt)
        acc[mt][nt] = wmma_bf16(fa[mt], fb, acc[mt][nt]);
    }

    if (hasNext) {
      uv4 p0, p1;
      cvt16(f0, f1, f2, f3, p0, p1);
      lds_store_32B(cur ? ldsB0 : ldsB1, p0, p1);
    }
    tile_waits();
    __syncthreads();
  }

  int lane  = t & 31;
  int col   = lane & 15;
  int rbase = (lane >> 4) << 3;
#pragma unroll
  for (int mt = 0; mt < 4; ++mt)
#pragma unroll
    for (int nt = 0; nt < 4; ++nt) {
      int n = nBase + wn * 64 + nt * 16 + col;
      float bias = b2[n];
#pragma unroll
      for (int v = 0; v < 8; ++v) {
        int m = mBase + wm * 64 + mt * 16 + rbase + v;
        out[(long)m * VOCAB + n] = acc[mt][nt][v] + bias;
      }
    }
}

extern "C" void kernel_launch(void* const* d_in, const int* in_sizes, int n_in,
                              void* d_out, int out_size, void* d_ws, size_t ws_size,
                              hipStream_t stream) {
  const int*   ids   = (const int*)d_in[0];
  const float* emb   = (const float*)d_in[1];
  const float* gamma = (const float*)d_in[2];
  const float* w1    = (const float*)d_in[3];
  const float* b1    = (const float*)d_in[4];
  const float* w2    = (const float*)d_in[5];
  const float* b2    = (const float*)d_in[6];
  float* out = (float*)d_out;

  // Workspace: x_bf16 [4096,1024] (8 MB) then y_bf16 [4096,4096] (32 MB)
  unsigned short* xb = (unsigned short*)d_ws;
  unsigned short* yb = xb + (size_t)NTOK * DMODEL;

  embed_rmsnorm_kernel<<<NTOK, 256, 0, stream>>>(ids, emb, gamma, xb);

  dim3 g1(NTOK / G1_BM, HIDDEN / G1_BN);   // 32 x 64
  gemm1_swiglu_kernel<<<g1, 256, 0, stream>>>(xb, w1, b1, yb);

  dim3 g2(NTOK / G2_BM, VOCAB / G2_BN);    // 16 x 250
  gemm2_kernel<<<g2, 256, 0, stream>>>(yb, w2, b2, out);
}
